// GroupQueryAttention_50672024158697
// MI455X (gfx1250) — matrix operand order repaired and checked by reading, not compiled, not run on hardware
//
#include <hip/hip_runtime.h>

// ---------------------------------------------------------------------------
// GQA (B=2, S=2048, 16 heads x hd=128, 4 KV groups) + output projection.
// MI455X / gfx1250: wave32, bf16 WMMA 16x16x32 with fp32 accumulation.
//
// Pipeline:
//   kernel 1: flash attention per head  -> d_ws as bf16 [4096 x 2048]
//   kernel 2: GEMM  out = X @ Wc^T + bc -> d_out fp32  [4096 x 2048]
//             (X tile staged into LDS by the Tensor Data Mover)
// ---------------------------------------------------------------------------

typedef __attribute__((ext_vector_type(16))) __bf16 v16bf;
typedef __attribute__((ext_vector_type(8)))  float  v8f;
typedef __attribute__((ext_vector_type(4)))  unsigned int u32x4;
typedef __attribute__((ext_vector_type(4)))  int    i32x4;
typedef __attribute__((ext_vector_type(8)))  int    i32x8;

#if defined(__HIP_DEVICE_COMPILE__) &&                         \
    __has_builtin(__builtin_amdgcn_tensor_load_to_lds) &&      \
    __has_builtin(__builtin_amdgcn_s_wait_tensorcnt)
#define HAVE_TDM 1
#else
#define HAVE_TDM 0
#endif

union FragBF {
  v16bf v;
  unsigned int u[8];
};

static __device__ __forceinline__ unsigned int pack2bf(float a, float b) {
  union { __bf16 h[2]; unsigned int u; } x;
  x.h[0] = (__bf16)a;
  x.h[1] = (__bf16)b;
  return x.u;
}

static __device__ __forceinline__ v8f zero8() {
  v8f z = {0.f, 0.f, 0.f, 0.f, 0.f, 0.f, 0.f, 0.f};
  return z;
}

static __device__ __forceinline__ v8f wmma_bf16(const FragBF& a, const FragBF& b, v8f c) {
  // (neg_a, A, neg_b, B, c_mod, C, reuse_a, reuse_b)
  return __builtin_amdgcn_wmma_f32_16x16x32_bf16(false, a.v, false, b.v, (short)0, c,
                                                 false, false);
}

#define S_LEN 2048
#define DM    2048
#define KVD   512
#define HD    128
#define QT    64      // query rows per block (4 waves x 16)
#define KT    32      // key tile
#define QS    136     // Qb/Kb row stride (bf16) -> 68 dwords/row, conflict-free frag loads
#define KS    136
#define VS    40      // V^T row stride (bf16)   -> 20 dwords/row
#define SCS   33      // score row stride (f32)
#define PS    40      // P row stride (bf16)

// ---------------------------------------------------------------------------
// Kernel 1: flash attention for one head / one 64-row query tile per block.
// ---------------------------------------------------------------------------
__global__ __launch_bounds__(128) void gqa_attn_bf16(
    const float* __restrict__ qg, const float* __restrict__ kg,
    const float* __restrict__ vg, __bf16* __restrict__ og) {
  __shared__ __bf16 Qb[QT * QS];        // 17.0 KB  Q tile, bf16 row-major
  __shared__ __bf16 Kb[KT * KS];        //  8.5 KB  K tile, bf16 row-major
  __shared__ __bf16 VT[HD * VS];        // 10.0 KB  V tile, transposed [hd][key]
  __shared__ float  sc[4 * 16 * SCS];   //  8.25 KB per-wave score tiles
  __shared__ __bf16 Pb[4 * 16 * PS];    //  5.0 KB  per-wave P tiles
  __shared__ float  cr[4 * 16];         //  0.25 KB per-row correction / 1/l

  const int t    = threadIdx.x;
  const int wave = t >> 5;
  const int lane = t & 31;
  const int half = lane >> 4;
  const int ln   = lane & 15;

  const int q0   = blockIdx.x * QT;
  const int head = blockIdx.y;          // = b*16 + (g*Hg + h)
  const int b    = head >> 4;
  const int gh   = head & 15;           // feature block index (g*Hg+h)
  const int grp  = gh >> 2;             // KV group

  const size_t qbase  = ((size_t)b * S_LEN + q0) * DM + (size_t)gh * HD;
  const size_t kvrow0 = (size_t)b * S_LEN * KVD + (size_t)grp * HD;
  const size_t obase  = ((size_t)b * S_LEN + q0 + wave * 16) * DM + (size_t)gh * HD;

  // ---- stage Q tile: 64x128 fp32 -> bf16 (2048 float4, 16 per thread) ----
  #pragma unroll
  for (int i = 0; i < 16; ++i) {
    int idx = t + i * 128;
    int r   = idx >> 5;
    int c4  = idx & 31;
    const float4 f = *(const float4*)(qg + qbase + (size_t)r * DM + c4 * 4);
    *(unsigned int*)&Qb[r * QS + c4 * 4 + 0] = pack2bf(f.x, f.y);
    *(unsigned int*)&Qb[r * QS + c4 * 4 + 2] = pack2bf(f.z, f.w);
  }
  __syncthreads();

  // ---- per-wave Q A-fragments (16x32 per k-chunk, 4 chunks over hd=128) ----
  FragBF qf[4];
  const int arow = wave * 16 + ln;
  #pragma unroll
  for (int c = 0; c < 4; ++c) {
    #pragma unroll
    for (int vv = 0; vv < 8; ++vv) {
      int kk = c * 32 + 2 * vv + half * 16;  // A layout (contiguous k = 16*half + s)
      qf[c].u[vv] = *(const unsigned int*)&Qb[arow * QS + kk];
    }
  }

  v8f Oacc[8];
  #pragma unroll
  for (int i = 0; i < 8; ++i) Oacc[i] = zero8();

  float mrun = -1e30f, lrun = 0.f;        // meaningful in lanes 0..15 (row = lane)
  const float scale = 0.08838834764831845f;  // 1/sqrt(128)

  for (int kt = 0; kt < S_LEN / KT; ++kt) {
    const size_t kvb = kvrow0 + (size_t)kt * KT * KVD;

    // ---- stage K (row-major) and V (transposed), fp32 -> bf16 ----
    #pragma unroll
    for (int i = 0; i < 8; ++i) {
      int idx = t + i * 128;
      int r   = idx >> 5;                 // key 0..31
      int c4  = idx & 31;                 // float4 within hd
      const float4 fk = *(const float4*)(kg + kvb + (size_t)r * KVD + c4 * 4);
      *(unsigned int*)&Kb[r * KS + c4 * 4 + 0] = pack2bf(fk.x, fk.y);
      *(unsigned int*)&Kb[r * KS + c4 * 4 + 2] = pack2bf(fk.z, fk.w);
      const float4 fv = *(const float4*)(vg + kvb + (size_t)r * KVD + c4 * 4);
      VT[(c4 * 4 + 0) * VS + r] = (__bf16)fv.x;
      VT[(c4 * 4 + 1) * VS + r] = (__bf16)fv.y;
      VT[(c4 * 4 + 2) * VS + r] = (__bf16)fv.z;
      VT[(c4 * 4 + 3) * VS + r] = (__bf16)fv.w;
    }
    // prefetch next K/V tile into cache (global_prefetch_b8)
    if (kt + 1 < S_LEN / KT) {
      const size_t nb = kvb + (size_t)KT * KVD;
      const int pr = t >> 2, pc = (t & 3) * 32;
      __builtin_prefetch(kg + nb + (size_t)pr * KVD + pc, 0, 1);
      __builtin_prefetch(vg + nb + (size_t)pr * KVD + pc, 0, 1);
    }
    __syncthreads();

    // ---- S = Q K^T : 16x32 scores per wave (8 WMMAs) ----
    v8f s0 = zero8(), s1 = zero8();
    #pragma unroll
    for (int c = 0; c < 4; ++c) {
      FragBF kb0, kb1;
      #pragma unroll
      for (int vv = 0; vv < 8; ++vv) {
        int kk = c * 32 + 2 * vv + half * 16;          // B layout (K stored [n][k])
        kb0.u[vv] = *(const unsigned int*)&Kb[(ln)      * KS + kk];
        kb1.u[vv] = *(const unsigned int*)&Kb[(16 + ln) * KS + kk];
      }
      s0 = wmma_bf16(qf[c], kb0, s0);
      s1 = wmma_bf16(qf[c], kb1, s1);
    }

    // ---- scatter scaled scores to LDS (C/D layout: row=half*8+r, col=ln) ----
    #pragma unroll
    for (int r = 0; r < 8; ++r) {
      sc[(wave * 16 + half * 8 + r) * SCS + ln]      = s0[r] * scale;
      sc[(wave * 16 + half * 8 + r) * SCS + 16 + ln] = s1[r] * scale;
    }
    __syncthreads();

    // ---- online softmax: lane r (<16) owns query row r of this wave ----
    if (lane < 16) {
      const int r = lane;
      float vbuf[32];
      float mt = mrun;
      #pragma unroll
      for (int j = 0; j < 32; ++j) {
        vbuf[j] = sc[(wave * 16 + r) * SCS + j];
        mt = fmaxf(mt, vbuf[j]);
      }
      const float c_ = __expf(mrun - mt);
      float ps = 0.f;
      #pragma unroll
      for (int j = 0; j < 32; ++j) {
        float p = __expf(vbuf[j] - mt);
        ps += p;
        Pb[(wave * 16 + r) * PS + j] = (__bf16)p;
      }
      lrun = lrun * c_ + ps;
      mrun = mt;
      cr[wave * 16 + r] = c_;
    }
    __syncthreads();

    // ---- rescale O by per-row correction, then O += P V (8 WMMAs) ----
    #pragma unroll
    for (int r = 0; r < 8; ++r) {
      const float c_ = cr[wave * 16 + half * 8 + r];
      #pragma unroll
      for (int nb = 0; nb < 8; ++nb) Oacc[nb][r] *= c_;
    }
    FragBF pf;
    #pragma unroll
    for (int vv = 0; vv < 8; ++vv) {
      int kk = 2 * vv + half * 16;   // A layout (contiguous k), K=key idx
      pf.u[vv] = *(const unsigned int*)&Pb[(wave * 16 + ln) * PS + kk];
    }
    #pragma unroll
    for (int nb = 0; nb < 8; ++nb) {
      FragBF vf;
      #pragma unroll
      for (int vv = 0; vv < 8; ++vv) {
        int kk = 2 * vv + half * 16;                        // B layout, V^T [hd][key]
        vf.u[vv] = *(const unsigned int*)&VT[(nb * 16 + ln) * VS + kk];
      }
      Oacc[nb] = wmma_bf16(pf, vf, Oacc[nb]);
    }
    __syncthreads();
  }

  // ---- normalize by 1/l and store bf16 to workspace ----
  if (lane < 16) cr[wave * 16 + lane] = 1.f / lrun;
  __syncthreads();
  #pragma unroll
  for (int r = 0; r < 8; ++r) {
    const float inv = cr[wave * 16 + half * 8 + r];
    #pragma unroll
    for (int nb = 0; nb < 8; ++nb) {
      og[obase + (size_t)(half * 8 + r) * DM + nb * 16 + ln] =
          (__bf16)(Oacc[nb][r] * inv);
    }
  }
}

// ---------------------------------------------------------------------------
// Kernel 2: out[m][n] = sum_k X[m][k] * Wc[n][k] + bc[n]
//   X: bf16 [4096 x 2048] (workspace), Wc/bc/out: fp32.
//   64x64 output tile per block, 4 waves, K chunks of 32.
//   X tile is DMA'd into LDS by the Tensor Data Mover: 2D descriptor,
//   data_size=2B, tile 32x64, row pad 4 dwords -> padded LDS stride XS=40.
// ---------------------------------------------------------------------------
#define XS 40
__global__ __launch_bounds__(128) void proj_bf16(
    const __bf16* __restrict__ X, const float* __restrict__ W,
    const float* __restrict__ bias, float* __restrict__ out) {
  __shared__ __bf16 Xb[64 * XS];
  __shared__ __bf16 Wb[64 * XS];

  const int t    = threadIdx.x;
  const int wave = t >> 5;
  const int lane = t & 31;
  const int half = lane >> 4;
  const int ln   = lane & 15;
  const int n0   = blockIdx.x * 64;
  const int m0   = blockIdx.y * 64;

  v8f acc[4];
  #pragma unroll
  for (int i = 0; i < 4; ++i) acc[i] = zero8();

  for (int kc = 0; kc < DM; kc += 32) {
#if HAVE_TDM
    // ---- X tile via TDM: one DMA issued by wave 0 ----
    if (wave == 0) {
      const unsigned lds_off = (unsigned)(size_t)(&Xb[0]);
      const unsigned long long ga =
          (unsigned long long)(size_t)X + ((size_t)m0 * DM + kc) * 2ull;
      u32x4 g0;
      g0.x = 1u;                                             // count=1, user mode
      g0.y = lds_off;                                        // lds_addr (bytes)
      g0.z = (unsigned)(ga & 0xFFFFFFFFull);                 // global_addr[31:0]
      g0.w = (unsigned)((ga >> 32) & 0x1FFFFFFull) | (2u << 30); // [56:32] | type=2
      i32x8 g1;
      g1[0] = (int)((1u << 16)      // data_size = 2 bytes
                  | (1u << 20)      // pad_enable
                  | (3u << 22)      // pad_interval: 16 dwords (one 32-bf16 row)
                  | (3u << 25));    // pad_amount:   4 dwords (8 bf16)
      g1[1] = (int)(((unsigned)DM & 0xFFFFu) << 16);          // tensor_dim0 lo16
      g1[2] = (int)(((unsigned)DM >> 16) |
                    (((unsigned)(2 * S_LEN) & 0xFFFFu) << 16)); // dim0 hi | dim1 lo
      g1[3] = (int)(((unsigned)(2 * S_LEN) >> 16) | (32u << 16)); // dim1 hi | tile_dim0
      g1[4] = 64;                                             // tile_dim1 (tile_dim2=0)
      g1[5] = DM;                                             // tensor_dim0_stride lo32
      g1[6] = 0;
      g1[7] = 0;
      i32x4 z4 = {0, 0, 0, 0};
#if __clang_major__ >= 23
      i32x8 z8 = {0, 0, 0, 0, 0, 0, 0, 0};
      __builtin_amdgcn_tensor_load_to_lds(g0, g1, z4, z4, z8, 0);
#else
      __builtin_amdgcn_tensor_load_to_lds(g0, g1, z4, z4, 0);
#endif
    }
#else
    // ---- fallback: X tile staged by vector loads ----
    #pragma unroll
    for (int i = 0; i < 8; ++i) {
      int idx = t + i * 128;
      int r   = idx >> 4;
      int cu  = idx & 15;
      *(unsigned int*)&Xb[r * XS + cu * 2] =
          *(const unsigned int*)(X + (size_t)(m0 + r) * DM + kc + cu * 2);
    }
#endif

    // ---- W tile: fp32 -> bf16 staging (all threads) ----
    #pragma unroll
    for (int i = 0; i < 8; ++i) {
      int idx = t + i * 128;
      int r   = idx >> 4;                 // tile row 0..63
      int cu  = idx & 15;                 // uint (=2 bf16) within 32-wide chunk
      const float2 f = *(const float2*)(W + (size_t)(n0 + r) * DM + kc + cu * 2);
      *(unsigned int*)&Wb[r * XS + cu * 2] = pack2bf(f.x, f.y);
    }

#if HAVE_TDM
    if (wave == 0) __builtin_amdgcn_s_wait_tensorcnt(0);
#endif
    __syncthreads();

    FragBF a;
    const int arow = wave * 16 + ln;
    #pragma unroll
    for (int vv = 0; vv < 8; ++vv) {
      int kk = 2 * vv + half * 16;   // A layout (contiguous k)
      a.u[vv] = *(const unsigned int*)&Xb[arow * XS + kk];
    }
    #pragma unroll
    for (int nb = 0; nb < 4; ++nb) {
      FragBF bfg;
      #pragma unroll
      for (int vv = 0; vv < 8; ++vv) {
        int kk = 2 * vv + half * 16;                        // B layout (Wc [n][k])
        bfg.u[vv] = *(const unsigned int*)&Wb[(nb * 16 + ln) * XS + kk];
      }
      acc[nb] = wmma_bf16(a, bfg, acc[nb]);
    }
    __syncthreads();
  }

  #pragma unroll
  for (int nb = 0; nb < 4; ++nb) {
    const int n  = n0 + nb * 16 + ln;
    const float bb = bias[n];
    #pragma unroll
    for (int r = 0; r < 8; ++r) {
      const int m = m0 + wave * 16 + half * 8 + r;
      out[(size_t)m * DM + n] = acc[nb][r] + bb;
    }
  }
}

// ---------------------------------------------------------------------------
extern "C" void kernel_launch(void* const* d_in, const int* in_sizes, int n_in,
                              void* d_out, int out_size, void* d_ws, size_t ws_size,
                              hipStream_t stream) {
  const float* q  = (const float*)d_in[0];
  const float* k  = (const float*)d_in[1];
  const float* v  = (const float*)d_in[2];
  const float* Wc = (const float*)d_in[3];
  const float* bc = (const float*)d_in[4];
  float* out = (float*)d_out;
  __bf16* attn_ws = (__bf16*)d_ws;   // 4096*2048 bf16 = 16 MB intermediate

  // kernel 1: 32 query tiles x (B*16 heads) blocks, 128 threads (4 waves)
  dim3 g1(S_LEN / QT, 2 * 16);
  gqa_attn_bf16<<<g1, 128, 0, stream>>>(q, k, v, attn_ws);

  // kernel 2: N/64 x M/64 blocks, 128 threads (4 waves)
  dim3 g2(DM / 64, (2 * S_LEN) / 64);
  proj_bf16<<<g2, 128, 0, stream>>>(attn_ws, Wc, bc, out);
}
